// SparseResidualBlock_25967372272143
// MI455X (gfx1250) — compile-verified, matrix operand order checked by reading, not run on hardware
//
#include <hip/hip_runtime.h>

typedef __attribute__((ext_vector_type(16))) __bf16 v16bf;
typedef __attribute__((ext_vector_type(8)))  float  v8f;
typedef __attribute__((ext_vector_type(4)))  __bf16 v4bf;

#define NPTS   2000000
#define KOFF   27
#define TILES  (NPTS / 16)        // 125000
#define CBLKS  (TILES / 8)        // 15625 blocks, 8 waves each
#define BNEPS  1e-4f

// workspace byte offsets
#define OFF_WB   0x0u             // bf16 weight blocks: 2*54*512 elems = 110592 B
#define OFF_ST   0x40000u         // 2 * 64 floats (scale[32], shift[32])
#define OFF_PART 0x41000u         // per-block partials: 15625*64*4 = 4,000,000 B
#define OFF_XB   0x800000u        // bf16 activation rows [N,32] (reused conv1-in & conv2-in)
#define OFF_R1   0x8800000u       // conv1 raw output, bf16 [N,32]

// ---------------- weight pre-conversion: f32 [27,32,32] -> bf16 B-operand blocks ----------
// Block (conv c, offset k, col-half h) = 32 lanes x 16 bf16, lane-major.
// Lane L holds column n=(L&15)+h*16; element e maps to K = e + (L>>4)*8 + (e>=8 ? 8 : 0).
__global__ void prep_weights(const float* __restrict__ W1, const float* __restrict__ W2,
                             __bf16* __restrict__ wb) {
  int tid = blockIdx.x * 256 + threadIdx.x;       // 2*54*512 = 55296 exact
  int e   = tid & 15;
  int L   = (tid >> 4) & 31;
  int b   = tid >> 9;                             // 0..107
  int c   = b / 54;
  int rem = b % 54;
  int k   = rem >> 1;
  int h   = rem & 1;
  int kq  = L >> 4;
  int n   = (L & 15) + h * 16;
  int K   = e + kq * 8 + ((e & 8) ? 8 : 0);
  const float* W = c ? W2 : W1;
  wb[tid] = (__bf16)W[(k * 32 + K) * 32 + n];
}

// ---------------- feat f32 -> bf16 rows ----------------
__global__ void feat_to_bf16(const float* __restrict__ f, __bf16* __restrict__ xb) {
  size_t t = (size_t)blockIdx.x * 256 + threadIdx.x;   // N*32/4 = 16,000,000 exact
  float4 v = ((const float4*)f)[t];
  v4bf o;
  o[0] = (__bf16)v.x; o[1] = (__bf16)v.y; o[2] = (__bf16)v.z; o[3] = (__bf16)v.w;
  ((v4bf*)xb)[t] = o;
}

// ---------------- gather-WMMA conv + deterministic stat partials ----------------
template <bool OUT_BF16>
__global__ __launch_bounds__(256)
void conv_kern(const __bf16* __restrict__ x, const int* __restrict__ nbr,
               const __bf16* __restrict__ wb, __bf16* __restrict__ outb,
               float* __restrict__ outf, float* __restrict__ partials) {
  __shared__ __align__(32) __bf16 wl[54 * 512];   // 54 KB of B operands
  __shared__ float4 part[256];

  // cooperative weight stage: 55296 B = 3456 uint4
  {
    const uint4* src = (const uint4*)wb;
    uint4* dst = (uint4*)wl;
    for (int i = threadIdx.x; i < 3456; i += 256) dst[i] = src[i];
  }
  __syncthreads();

  const int l    = threadIdx.x & 31;
  const int wv   = threadIdx.x >> 5;
  const int tile = blockIdx.x * 8 + wv;           // 125000 tiles exact, EXEC stays full
  const int half = l >> 4;
  const int r    = tile * 16 + (l & 15);
  const int co0  = half * 8;                      // bf16 element offsets of the two K-chunks
  const int co1  = 16 + half * 8;

  int nb[KOFF];
  {
    const int* np = nbr + (size_t)r * KOFF;
#pragma unroll
    for (int k = 0; k < KOFF; ++k) nb[k] = np[k];
  }

  v8f acc0 = {};
  v8f acc1 = {};
  for (int k = 0; k < KOFF; ++k) {
    const __bf16* xp = x + (size_t)nb[k] * 32;
    union { uint4 u[2]; v16bf v; } A;
    A.u[0] = *(const uint4*)(xp + co0);
    A.u[1] = *(const uint4*)(xp + co1);
    v16bf B0 = *(const v16bf*)(wl + (k * 2 + 0) * 512 + l * 16);
    v16bf B1 = *(const v16bf*)(wl + (k * 2 + 1) * 512 + l * 16);
    acc0 = __builtin_amdgcn_wmma_f32_16x16x32_bf16(false, A.v, false, B0, (short)0, acc0, false, false);
    acc1 = __builtin_amdgcn_wmma_f32_16x16x32_bf16(false, A.v, false, B1, (short)0, acc1, false, false);
  }

  // per-lane stat partials (channel l&15 from acc0, channel 16+(l&15) from acc1)
  float s0 = 0.f, q0 = 0.f, s1 = 0.f, q1 = 0.f;
#pragma unroll
  for (int v = 0; v < 8; ++v) {
    s0 += acc0[v]; q0 += acc0[v] * acc0[v];
    s1 += acc1[v]; q1 += acc1[v] * acc1[v];
  }
  part[threadIdx.x] = make_float4(s0, q0, s1, q1);

  // write the 16x32 output tile (C/D layout: VGPR v -> row v (lanes<16) / v+8 (lanes>=16))
  const int col = l & 15;
#pragma unroll
  for (int v = 0; v < 8; ++v) {
    const int row = tile * 16 + ((l < 16) ? v : v + 8);
    if (OUT_BF16) {
      __bf16* orow = outb + (size_t)row * 32;
      orow[col]      = (__bf16)acc0[v];
      orow[16 + col] = (__bf16)acc1[v];
    } else {
      float* orow = outf + (size_t)row * 32;
      orow[col]      = acc0[v];
      orow[16 + col] = acc1[v];
    }
  }

  __syncthreads();
  if (threadIdx.x < 16) {   // fixed-order block reduction -> deterministic
    const int t = threadIdx.x;
    float S0 = 0.f, Q0 = 0.f, S1 = 0.f, Q1 = 0.f;
#pragma unroll
    for (int ww = 0; ww < 8; ++ww) {
      float4 a = part[ww * 32 + t];
      float4 b = part[ww * 32 + t + 16];
      S0 += a.x + b.x; Q0 += a.y + b.y;
      S1 += a.z + b.z; Q1 += a.w + b.w;
    }
    float* p = partials + (size_t)blockIdx.x * 64;
    p[t] = S0; p[16 + t] = S1; p[32 + t] = Q0; p[48 + t] = Q1;
  }
}

// ---------------- fixed-order cross-block reduction + BN coefficient finalize ----------------
__global__ void reduce_finalize(const float* __restrict__ partials, int G,
                                const float* __restrict__ gamma, const float* __restrict__ beta,
                                float* __restrict__ st) {
  __shared__ float rs[256], rq[256];
  const int ch = blockIdx.x;      // 0..31
  float s = 0.f, q = 0.f;
  for (int g = threadIdx.x; g < G; g += 256) {
    s += partials[(size_t)g * 64 + ch];
    q += partials[(size_t)g * 64 + 32 + ch];
  }
  rs[threadIdx.x] = s; rq[threadIdx.x] = q;
  __syncthreads();
  for (int off = 128; off > 0; off >>= 1) {
    if (threadIdx.x < off) {
      rs[threadIdx.x] += rs[threadIdx.x + off];
      rq[threadIdx.x] += rq[threadIdx.x + off];
    }
    __syncthreads();
  }
  if (threadIdx.x == 0) {
    float mean = rs[0] / (float)NPTS;
    float var  = rq[0] / (float)NPTS - mean * mean;
    float sc   = gamma[ch] * rsqrtf(var + BNEPS);
    st[ch]      = sc;
    st[32 + ch] = beta[ch] - mean * sc;
  }
}

// ---------------- BN+ReLU applied to conv1 raw, emits bf16 rows for conv2 ----------------
__global__ void bnrelu_bf16(const __bf16* __restrict__ raw, const float* __restrict__ st,
                            __bf16* __restrict__ x2) {
  size_t t = (size_t)blockIdx.x * 256 + threadIdx.x;   // 16M exact
  int ch4 = (int)(t & 7) * 4;
  float4 s = *(const float4*)(st + ch4);
  float4 b = *(const float4*)(st + 32 + ch4);
  v4bf v = ((const v4bf*)raw)[t];
  v4bf o;
  o[0] = (__bf16)fmaxf((float)v[0] * s.x + b.x, 0.f);
  o[1] = (__bf16)fmaxf((float)v[1] * s.y + b.y, 0.f);
  o[2] = (__bf16)fmaxf((float)v[2] * s.z + b.z, 0.f);
  o[3] = (__bf16)fmaxf((float)v[3] * s.w + b.w, 0.f);
  ((v4bf*)x2)[t] = o;
}

// ---------------- final BN + residual + ReLU, in place on d_out ----------------
__global__ void bn_add_relu(float* __restrict__ out, const float* __restrict__ feat,
                            const float* __restrict__ st) {
  size_t t = (size_t)blockIdx.x * 256 + threadIdx.x;   // 16M exact
  int ch4 = (int)(t & 7) * 4;
  float4 s = *(const float4*)(st + ch4);
  float4 b = *(const float4*)(st + 32 + ch4);
  float4 r = ((const float4*)out)[t];
  float4 f = ((const float4*)feat)[t];
  float4 y;
  y.x = fmaxf(r.x * s.x + b.x + f.x, 0.f);
  y.y = fmaxf(r.y * s.y + b.y + f.y, 0.f);
  y.z = fmaxf(r.z * s.z + b.z + f.z, 0.f);
  y.w = fmaxf(r.w * s.w + b.w + f.w, 0.f);
  ((float4*)out)[t] = y;
}

extern "C" void kernel_launch(void* const* d_in, const int* in_sizes, int n_in,
                              void* d_out, int out_size, void* d_ws, size_t ws_size,
                              hipStream_t stream) {
  const float* feat = (const float*)d_in[0];
  const int*   nbr  = (const int*)d_in[1];
  const float* W1   = (const float*)d_in[2];
  const float* g1   = (const float*)d_in[3];
  const float* b1   = (const float*)d_in[4];
  const float* W2   = (const float*)d_in[5];
  const float* g2   = (const float*)d_in[6];
  const float* b2   = (const float*)d_in[7];
  float* out = (float*)d_out;

  char* ws = (char*)d_ws;
  __bf16* wb    = (__bf16*)(ws + OFF_WB);
  float*  st0   = (float*)(ws + OFF_ST);
  float*  st1   = st0 + 64;
  float*  parts = (float*)(ws + OFF_PART);
  __bf16* xb    = (__bf16*)(ws + OFF_XB);   // conv1 input; reused as conv2 input
  __bf16* raw1  = (__bf16*)(ws + OFF_R1);

  const int vecBlocks = (NPTS * 32 / 4) / 256;   // 62500

  prep_weights<<<216, 256, 0, stream>>>(W1, W2, wb);
  feat_to_bf16<<<vecBlocks, 256, 0, stream>>>(feat, xb);

  conv_kern<true><<<CBLKS, 256, 0, stream>>>(xb, nbr, wb, raw1, nullptr, parts);
  reduce_finalize<<<32, 256, 0, stream>>>(parts, CBLKS, g1, b1, st0);
  bnrelu_bf16<<<vecBlocks, 256, 0, stream>>>(raw1, st0, xb);

  conv_kern<false><<<CBLKS, 256, 0, stream>>>(xb, nbr, wb + 54 * 512, nullptr, out, parts);
  reduce_finalize<<<32, 256, 0, stream>>>(parts, CBLKS, g2, b2, st1);
  bn_add_relu<<<vecBlocks, 256, 0, stream>>>(out, feat, st1);
}
